// TemplatePointwiseAttention_33681133535515
// MI455X (gfx1250) — matrix operand order repaired and checked by reading, not compiled, not run on hardware
//
#include <hip/hip_runtime.h>
#include <hip/hip_bf16.h>

// Problem constants (B=1)
#define NRES   512
#define NPIX   (NRES*NRES)          // 262144 pixels
#define S_TPL  4
#define CT     128                  // c_t
#define CZ     64                   // c_z
#define NHEAD  4
#define TILES  (NPIX/16)            // 16384 tiles of 16 pixels

#define NW         4                // waves per block
#define THREADS    (NW*32)
#define BLOCKS     1024
#define WAVESLOTS  (BLOCKS*NW)

// exp(x) = exp2(x * log2(e)); fold the C_H^-0.5 = 0.25 score scale in too.
#define LOG2E      1.4426950408889634f
#define SCORE_L2   (0.25f * LOG2E)

typedef __attribute__((ext_vector_type(16))) _Float16 v16h;
typedef __attribute__((ext_vector_type(8)))  float    v8f;

// B-fragment slots in LDS (each fragment: 32 lanes * 16 halfs = 1KB)
#define FQ(kt,nt)  ((kt)*4+(nt))          // Wq: kt 0..3, nt 0..3 -> 0..15
#define FK(kt,nt)  (16+(kt)*4+(nt))       // Wk: kt 0..1, nt 0..3 -> 16..23
#define FV(kt,nt)  (24+(kt)*4+(nt))       // Wv: kt 0..1, nt 0..3 -> 24..31
#define FOW(kt,nt) (32+(kt)*8+(nt))       // Wo: kt 0..1, nt 0..7 -> 32..47
#define NFRAG 48

__device__ __forceinline__ v8f wmma_f16(v16h a, v16h b, v8f c) {
  // D = A(16x32 f16) * B(32x16 f16) + C(16x16 f32)
  return __builtin_amdgcn_wmma_f32_16x16x32_f16(
      /*neg_a=*/false, a, /*neg_b=*/false, b,
      /*c_mod=*/(short)0, c, /*reuse_a=*/false, /*reuse_b=*/false);
}

// Build a 16x32 f16 A-fragment from a row-major f32 tile in global memory.
// ISA layout: lanes 0-15 row M=lane, K {0..7,16..23}; lanes 16-31 row M=lane-16,
// K {8..15,24..31} (2 halfs per VGPR).
__device__ __forceinline__ v16h a_frag_global(const float* base, int rowStride,
                                              int kbase, int l15, int hi) {
  const float* p = base + (size_t)l15 * rowStride + kbase + hi * 8;
  v16h a;
#pragma unroll
  for (int i = 0; i < 8; ++i) a[i]     = (_Float16)p[i];       // global_load_b128 x2
#pragma unroll
  for (int i = 0; i < 8; ++i) a[8 + i] = (_Float16)p[16 + i];  // global_load_b128 x2
  return a;
}

__global__ __launch_bounds__(THREADS)
void tpa_fused_kernel(const float* __restrict__ t, const float* __restrict__ z,
                      const float* __restrict__ tmask,
                      const float* __restrict__ Wq, const float* __restrict__ Wk,
                      const float* __restrict__ Wv, const float* __restrict__ Wo,
                      const float* __restrict__ bo, float* __restrict__ out) {
  __shared__ _Float16 wfrag[NFRAG * 512];   // 48 KB: weight B-fragments, f16
  __shared__ _Float16 obuf[NW * 16 * 64];   // 8 KB: per-wave o re-layout buffer

  const int lane = threadIdx.x & 31;
  const int wave = threadIdx.x >> 5;
  const int l15  = lane & 15;
  const int hi   = (lane >> 4) & 1;

  // ---- Stage weight B-fragments into LDS once per block (f32 -> f16) ----
  // B layout (32x16 f16): same K order as A: lanes 0-15: K {0..7,16..23}, N=lane;
  // lanes 16-31: K {8..15,24..31}, N=lane-16.
  for (int f = wave; f < NFRAG; f += NW) {
    const float* W; int Nw, kb, nb;
    if (f < 16)      { W = Wq; Nw = 64;  kb = (f >> 2) * 32;        nb = (f & 3) * 16; }
    else if (f < 24) { W = Wk; Nw = 64;  kb = ((f - 16) >> 2) * 32; nb = ((f - 16) & 3) * 16; }
    else if (f < 32) { W = Wv; Nw = 64;  kb = ((f - 24) >> 2) * 32; nb = ((f - 24) & 3) * 16; }
    else             { W = Wo; Nw = 128; kb = ((f - 32) >> 3) * 32; nb = ((f - 32) & 7) * 16; }
    v16h fr;
#pragma unroll
    for (int i = 0; i < 16; ++i) {
      int r = i >> 1;
      int k = kb + (i >> 3) * 16 + hi * 8 + (i & 7);
      fr[i] = (_Float16)W[k * Nw + nb + l15];
    }
    *(v16h*)(wfrag + f * 512 + lane * 16) = fr;
  }
  __syncthreads();

  _Float16* ob = obuf + wave * 16 * 64;
  const int stride = WAVESLOTS;

  for (int tile = blockIdx.x * NW + wave; tile < TILES; tile += stride) {
    const int p0 = tile * 16;

    // Prefetch next tile's inputs into cache (global_prefetch_b8)
    {
      int nx = tile + stride;
      if (nx < TILES) {
        __builtin_prefetch(z + (size_t)nx * 16 * CT, 0, 0);
        __builtin_prefetch(t + (size_t)nx * 16 * CZ, 0, 0);
      }
    }

    // ---- q = z_tile(16x128) @ Wq(128x64), per-head D fragments ----
    const float* zr = z + (size_t)p0 * CT;
    v8f qf[NHEAD];
    {
      v16h az[4];
#pragma unroll
      for (int kt = 0; kt < 4; ++kt) az[kt] = a_frag_global(zr, CT, kt * 32, l15, hi);
#pragma unroll
      for (int h = 0; h < NHEAD; ++h) {
        v8f acc = {};
#pragma unroll
        for (int kt = 0; kt < 4; ++kt) {
          v16h b = *(const v16h*)(wfrag + FQ(kt, h) * 512 + lane * 16);
          acc = wmma_f16(az[kt], b, acc);
        }
        qf[h] = acc;
      }
    }

    // ---- Online softmax-weighted sum over templates (registers only) ----
    // Keep the s-loop ROLLED: one template's fragments live at a time ->
    // VGPR usage stays < 256 (no s_set_vgpr_msb), occupancy does the
    // latency hiding for this HBM-bound kernel.
    v8f oacc[NHEAD], den[NHEAD];
#pragma unroll
    for (int h = 0; h < NHEAD; ++h) { oacc[h] = {}; den[h] = {}; }

#pragma unroll 1
    for (int s = 0; s < S_TPL; ++s) {
      // mask bias, already in log2 space (uniform scalar load, K$ resident)
      const float bias_l2 = (1e9f * LOG2E) * (tmask[s] - 1.0f);

      const float* tr = t + ((size_t)s * NPIX + p0) * CZ;
      v16h at2[2];
#pragma unroll
      for (int kt = 0; kt < 2; ++kt) at2[kt] = a_frag_global(tr, CZ, kt * 32, l15, hi);

#pragma unroll
      for (int h = 0; h < NHEAD; ++h) {
        v8f kf = {}, vf = {};
#pragma unroll
        for (int kt = 0; kt < 2; ++kt) {
          v16h bk = *(const v16h*)(wfrag + FK(kt, h) * 512 + lane * 16);
          kf = wmma_f16(at2[kt], bk, kf);
        }
#pragma unroll
        for (int kt = 0; kt < 2; ++kt) {
          v16h bv = *(const v16h*)(wfrag + FV(kt, h) * 512 + lane * 16);
          vf = wmma_f16(at2[kt], bv, vf);
        }
        // score[p][h] = sum_c q*k : c lives across lanes 0-15 / 16-31 of the
        // D fragment -> 4-step butterfly leaves the sum broadcast per half,
        // which is exactly the layout needed to scale the v fragment.
#pragma unroll
        for (int e = 0; e < 8; ++e) {
          float x = qf[h][e] * kf[e];
          x += __shfl_xor(x, 1);
          x += __shfl_xor(x, 2);
          x += __shfl_xor(x, 4);
          x += __shfl_xor(x, 8);
          float ex = __builtin_amdgcn_exp2f(x * SCORE_L2 + bias_l2);
          den[h][e]  += ex;
          oacc[h][e] += ex * vf[e];
        }
      }
    }

    // ---- Normalize; re-layout D-frag -> A-frag via per-wave LDS ----
#pragma unroll
    for (int h = 0; h < NHEAD; ++h) {
#pragma unroll
      for (int r = 0; r < 8; ++r) {
        float v = oacc[h][r] / den[h][r];
        ob[(r + 8 * hi) * 64 + h * 16 + l15] = (_Float16)v;   // row p, ch h*16+c
      }
    }
    // same-wave LDS RAW: compiler inserts s_wait_dscnt

    v16h ao[2];
#pragma unroll
    for (int kt = 0; kt < 2; ++kt) {
      const _Float16* p = ob + l15 * 64 + kt * 32 + hi * 8;
      v16h a;
#pragma unroll
      for (int i = 0; i < 8; ++i) a[i]     = p[i];        // ds_load_b128
#pragma unroll
      for (int i = 0; i < 8; ++i) a[8 + i] = p[16 + i];   // ds_load_b128
      ao[kt] = a;
    }

    // ---- out = o(16x64) @ Wo(64x128) + bo ----
#pragma unroll
    for (int n = 0; n < 8; ++n) {
      v8f acc = {};
#pragma unroll
      for (int kt = 0; kt < 2; ++kt) {
        v16h b = *(const v16h*)(wfrag + FOW(kt, n) * 512 + lane * 16);
        acc = wmma_f16(ao[kt], b, acc);
      }
      const int ch = n * 16 + l15;
      const float bb = bo[ch];
#pragma unroll
      for (int r = 0; r < 8; ++r) {
        out[(size_t)(p0 + r + 8 * hi) * CT + ch] = acc[r] + bb;
      }
    }
  }
}

extern "C" void kernel_launch(void* const* d_in, const int* in_sizes, int n_in,
                              void* d_out, int out_size, void* d_ws, size_t ws_size,
                              hipStream_t stream) {
  // setup_inputs order: t, z, template_mask, Wq, Wk, Wv, Wo, bo  (all f32)
  const float* t  = (const float*)d_in[0];
  const float* z  = (const float*)d_in[1];
  const float* mk = (const float*)d_in[2];
  const float* Wq = (const float*)d_in[3];
  const float* Wk = (const float*)d_in[4];
  const float* Wv = (const float*)d_in[5];
  const float* Wo = (const float*)d_in[6];
  const float* bo = (const float*)d_in[7];
  float* out = (float*)d_out;

  hipLaunchKernelGGL(tpa_fused_kernel, dim3(BLOCKS), dim3(THREADS), 0, stream,
                     t, z, mk, Wq, Wk, Wv, Wo, bo, out);
}
